// timeMiningBlock_17051020165916
// MI455X (gfx1250) — matrix operand order repaired and checked by reading, not compiled, not run on hardware
//
#include <hip/hip_runtime.h>
#include <hip/hip_bf16.h>

// ---------------------------------------------------------------------------
// CDNA5 (gfx1250) transformer decode-step:
//   out = x; out[:, -1, :] = MHA+MLP block applied to last position
// Dominant cost: kv = LN(past) @ Wkv  (32760 x 768 x 1536) -> bf16 WMMA GEMM.
// v2: each wave computes a 16x32 tile (2 accumulators sharing one A stream).
// ---------------------------------------------------------------------------

typedef __attribute__((ext_vector_type(16))) __bf16 v16bf;
typedef __attribute__((ext_vector_type(8)))  float  v8f;

#define LN_EPS 1e-6f

__device__ __forceinline__ unsigned short f2bf(float f) {
    unsigned u = __float_as_uint(f);
    unsigned r = 0x7FFFu + ((u >> 16) & 1u);   // round-to-nearest-even
    return (unsigned short)((u + r) >> 16);
}
__device__ __forceinline__ float bf2f(unsigned short s) {
    return __uint_as_float(((unsigned)s) << 16);
}

union FragBF16 { v16bf v; uint4 q[2]; };

// ---------------------------------------------------------------------------
// 0) Stream-copy x -> out (last row overwritten later in-stream).
// ---------------------------------------------------------------------------
__global__ void __launch_bounds__(256) copy_f4_kernel(const float4* __restrict__ s,
                                                      float4* __restrict__ d, long n) {
    long i = (long)blockIdx.x * 256 + threadIdx.x;
    if (i < n) d[i] = s[i];
}

// ---------------------------------------------------------------------------
// 1) Row LayerNorm of all 8*4096 rows of x -> bf16 xn.  One wave32 per row.
// ---------------------------------------------------------------------------
__global__ void __launch_bounds__(256) ln_rows_kernel(const float* __restrict__ x,
                                                      unsigned short* __restrict__ xn) {
    int wave = threadIdx.x >> 5;
    int lane = threadIdx.x & 31;
    long row = (long)blockIdx.x * 8 + wave;            // 32768 rows total
    const float* xr = x + row * 768;

    float vals[24];
    float s = 0.f;
#pragma unroll
    for (int i = 0; i < 24; ++i) { vals[i] = xr[lane + i * 32]; s += vals[i]; }
#pragma unroll
    for (int m = 16; m >= 1; m >>= 1) s += __shfl_xor(s, m, 32);
    float mean = s * (1.f / 768.f);

    float vs = 0.f;
#pragma unroll
    for (int i = 0; i < 24; ++i) { float d = vals[i] - mean; vs += d * d; }
#pragma unroll
    for (int m = 16; m >= 1; m >>= 1) vs += __shfl_xor(vs, m, 32);
    float inv = rsqrtf(vs * (1.f / 768.f) + LN_EPS);

    unsigned short* o = xn + row * 768;
#pragma unroll
    for (int i = 0; i < 24; ++i) o[lane + i * 32] = f2bf((vals[i] - mean) * inv);
}

// ---------------------------------------------------------------------------
// 2) Wkv (768x1536 f32) -> WkvT (1536x768 bf16) so B-fragment K is contiguous.
// ---------------------------------------------------------------------------
__global__ void __launch_bounds__(256) wkvT_bf16_kernel(const float* __restrict__ Wkv,
                                                        unsigned short* __restrict__ wkvT) {
    int idx = blockIdx.x * 256 + threadIdx.x;
    if (idx < 768 * 1536) {
        int k = idx / 1536, n = idx % 1536;
        wkvT[(size_t)n * 768 + k] = f2bf(Wkv[idx]);
    }
}

// ---------------------------------------------------------------------------
// 3) q = xn[:, -1, :] @ Wq + bq   (8 x 768)  -- tiny, one block per batch row.
// ---------------------------------------------------------------------------
__global__ void __launch_bounds__(256) q_proj_kernel(const unsigned short* __restrict__ xn,
                                                     const float* __restrict__ Wq,
                                                     const float* __restrict__ bq,
                                                     float* __restrict__ qbuf) {
    __shared__ float r[768];
    int b = blockIdx.x;
    for (int i = threadIdx.x; i < 768; i += 256)
        r[i] = bf2f(xn[((size_t)(b * 4096 + 4095)) * 768 + i]);
    __syncthreads();
#pragma unroll
    for (int cc = 0; cc < 3; ++cc) {
        int c = threadIdx.x + cc * 256;
        float acc = bq[c];
        for (int k = 0; k < 768; ++k) acc += r[k] * Wq[k * 768 + c];
        qbuf[b * 768 + c] = acc;
    }
}

// ---------------------------------------------------------------------------
// 4) KV GEMM: kv[M=32760, 1536] = xn_past @ Wkv + bkv,  bf16 WMMA.
//    Block = 256 threads (8 waves) -> 16 x 256 output tile.
//    Wave  -> 16x32 tile: one A stream feeds 2 independent accumulators,
//    2 x 24 v_wmma_f32_16x16x32_bf16 per wave.
// ---------------------------------------------------------------------------
__global__ void __launch_bounds__(256) kv_gemm_kernel(const unsigned short* __restrict__ xn,
                                                      const unsigned short* __restrict__ wkvT,
                                                      const float* __restrict__ bkv,
                                                      unsigned short* __restrict__ kv) {
    __shared__ unsigned short atile[16 * 768];         // 24 KB: 16 M-rows x K=768 bf16
    const int M = 8 * 4095;
    int mbase = blockIdx.y * 16;

    // Stage A-panel into LDS: 1536 x uint4 (16B each), 6 per thread.
    for (int i = threadIdx.x; i < 1536; i += 256) {
        int r = i / 96;                                // tile row
        int w = i % 96;                                // uint4 within row
        int m = mbase + r;
        if (m >= M) m = M - 1;                         // clamp ragged tail (unused)
        int b = m / 4095, n = m % 4095;                // past rows only
        const uint4* src = (const uint4*)(xn + ((size_t)(b * 4096 + n)) * 768);
        ((uint4*)atile)[i] = src[w];
    }
    __syncthreads();

    int wave = threadIdx.x >> 5;
    int lane = threadIdx.x & 31;
    int nbase = blockIdx.x * 256 + wave * 32;          // wave's 32-column strip
    int col0  = nbase + (lane & 15);
    int col1  = col0 + 16;
    int khalf = lane >> 4;                             // 0: low K-half, 1: high

    // A fragment (16-bit A 16x32, ISA 7.12.2): lane holds row (lane&15),
    // K = [khalf*8, khalf*8+8) in v0-3 and +16 in v4-7.
    const unsigned short* albase = atile + (lane & 15) * 768 + khalf * 8;
    // B fragment (32x16): lane holds column, K = [khalf*8, khalf*8+8) and +16 (same as A).
    const unsigned short* brow0 = wkvT + (size_t)col0 * 768 + khalf * 8;
    const unsigned short* brow1 = wkvT + (size_t)col1 * 768 + khalf * 8;

    v8f c0 = {};
    v8f c1 = {};
    for (int k = 0; k < 768; k += 32) {
        FragBF16 a, b0, b1;
        a.q[0]  = *(const uint4*)(albase + k);         // K = k+khalf*8 .. +8
        a.q[1]  = *(const uint4*)(albase + k + 16);    // K = +16
        b0.q[0] = *(const uint4*)(brow0 + k);          // K = k+khalf*8 .. +8
        b0.q[1] = *(const uint4*)(brow0 + k + 16);     // K = +16
        b1.q[0] = *(const uint4*)(brow1 + k);
        b1.q[1] = *(const uint4*)(brow1 + k + 16);
        c0 = __builtin_amdgcn_wmma_f32_16x16x32_bf16(
                 false, a.v, false, b0.v, (short)0, c0, false, false);
        c1 = __builtin_amdgcn_wmma_f32_16x16x32_bf16(
                 false, a.v, false, b1.v, (short)0, c1, false, false);
    }

    // C/D layout: VGPR r holds M = r + khalf*8, N = lane&15.
    float bias0 = bkv[col0];
    float bias1 = bkv[col1];
#pragma unroll
    for (int r = 0; r < 8; ++r) {
        int gm = mbase + r + khalf * 8;
        if (gm < M) {
            kv[(size_t)gm * 1536 + col0] = f2bf(c0[r] + bias0);
            kv[(size_t)gm * 1536 + col1] = f2bf(c1[r] + bias1);
        }
    }
}

// ---------------------------------------------------------------------------
// 5) Attention: one block per (b, head). sim -> softmax -> weighted V.
// ---------------------------------------------------------------------------
__global__ void __launch_bounds__(256) attn_kernel(const float* __restrict__ qbuf,
                                                   const unsigned short* __restrict__ kv,
                                                   float* __restrict__ attn_out) {
    __shared__ float qs[48];
    __shared__ float es[4096];
    __shared__ float red[256];
    __shared__ float part[5][48];

    int bh = blockIdx.x;
    int b = bh >> 4, h = bh & 15;
    int t = threadIdx.x;
    if (t < 48) qs[t] = qbuf[b * 768 + h * 48 + t];
    __syncthreads();

    const float scale = 0.14433756729740643f;          // 48^-0.5
    float lmax = -1e30f;
    for (int j = t; j < 4095; j += 256) {
        const unsigned short* kr = kv + ((size_t)(b * 4095 + j)) * 1536 + h * 48;
        float s = 0.f;
#pragma unroll
        for (int u = 0; u < 6; ++u) {
            uint4 pk = *(const uint4*)(kr + u * 8);
            const unsigned short* ps = (const unsigned short*)&pk;
#pragma unroll
            for (int e = 0; e < 8; ++e) s += bf2f(ps[e]) * qs[u * 8 + e];
        }
        s *= scale;
        es[j] = s;
        lmax = fmaxf(lmax, s);
    }
    red[t] = lmax; __syncthreads();
    for (int off = 128; off >= 1; off >>= 1) {
        if (t < off) red[t] = fmaxf(red[t], red[t + off]);
        __syncthreads();
    }
    float mx = red[0];
    __syncthreads();

    float lsum = 0.f;
    for (int j = t; j < 4095; j += 256) {
        float e = __expf(es[j] - mx);
        es[j] = e;
        lsum += e;
    }
    red[t] = lsum; __syncthreads();
    for (int off = 128; off >= 1; off >>= 1) {
        if (t < off) red[t] += red[t + off];
        __syncthreads();
    }
    float inv = 1.f / red[0];

    if (t < 240) {                                     // 5 key-chunks x 48 dims
        int d = t % 48, chunk = t / 48;
        int j0 = chunk * 819;
        int j1 = (chunk == 4) ? 4095 : (j0 + 819);
        float acc = 0.f;
        for (int j = j0; j < j1; ++j)
            acc += es[j] * bf2f(kv[((size_t)(b * 4095 + j)) * 1536 + 768 + h * 48 + d]);
        part[chunk][d] = acc;
    }
    __syncthreads();
    if (t < 48) {
        float a = part[0][t] + part[1][t] + part[2][t] + part[3][t] + part[4][t];
        attn_out[b * 768 + h * 48 + t] = a * inv;
    }
}

// ---------------------------------------------------------------------------
// 6) o = attn_out@Wo + bo; last = x_last + o; ln2 = LN(last).
// ---------------------------------------------------------------------------
__global__ void __launch_bounds__(256) proj_ln_kernel(const float* __restrict__ attn_out,
                                                      const float* __restrict__ Wo,
                                                      const float* __restrict__ bo,
                                                      const float* __restrict__ x,
                                                      float* __restrict__ last,
                                                      float* __restrict__ ln2) {
    __shared__ float ao[768];
    __shared__ float red[256];
    int b = blockIdx.x, t = threadIdx.x;
    for (int i = t; i < 768; i += 256) ao[i] = attn_out[b * 768 + i];
    __syncthreads();

    const float* xl = x + ((size_t)b * 4096 + 4095) * 768;
    float lv[3];
#pragma unroll
    for (int cc = 0; cc < 3; ++cc) {
        int c = t + cc * 256;
        float acc = bo[c];
        for (int k = 0; k < 768; ++k) acc += ao[k] * Wo[k * 768 + c];
        float l = xl[c] + acc;
        lv[cc] = l;
        last[b * 768 + c] = l;
    }
    float s = lv[0] + lv[1] + lv[2];
    red[t] = s; __syncthreads();
    for (int off = 128; off >= 1; off >>= 1) {
        if (t < off) red[t] += red[t + off];
        __syncthreads();
    }
    float mean = red[0] * (1.f / 768.f);
    __syncthreads();

    float vs = 0.f;
#pragma unroll
    for (int cc = 0; cc < 3; ++cc) { float d = lv[cc] - mean; vs += d * d; }
    red[t] = vs; __syncthreads();
    for (int off = 128; off >= 1; off >>= 1) {
        if (t < off) red[t] += red[t + off];
        __syncthreads();
    }
    float inv = rsqrtf(red[0] * (1.f / 768.f) + LN_EPS);
#pragma unroll
    for (int cc = 0; cc < 3; ++cc) {
        int c = t + cc * 256;
        ln2[b * 768 + c] = (lv[cc] - mean) * inv;
    }
}

// ---------------------------------------------------------------------------
// 7) g = gelu(ln2 @ W1 + b1)   (8 x 3072)
// ---------------------------------------------------------------------------
__global__ void __launch_bounds__(256) mlp1_kernel(const float* __restrict__ ln2,
                                                   const float* __restrict__ W1,
                                                   const float* __restrict__ b1,
                                                   float* __restrict__ g) {
    __shared__ float r[768];
    int b = blockIdx.x / 12, cc = blockIdx.x % 12;
    int c = cc * 256 + threadIdx.x;
    for (int i = threadIdx.x; i < 768; i += 256) r[i] = ln2[b * 768 + i];
    __syncthreads();
    float acc = b1[c];
    for (int k = 0; k < 768; ++k) acc += r[k] * W1[k * 3072 + c];
    float xv = acc;
    float tt = 0.7978845608028654f * (xv + 0.044715f * xv * xv * xv);
    g[b * 3072 + c] = 0.5f * xv * (1.f + tanhf(tt));
}

// ---------------------------------------------------------------------------
// 8) final = last + g @ W2 + b2 -> out[:, -1, :]
// ---------------------------------------------------------------------------
__global__ void __launch_bounds__(256) mlp2_kernel(const float* __restrict__ g,
                                                   const float* __restrict__ W2,
                                                   const float* __restrict__ b2,
                                                   const float* __restrict__ last,
                                                   float* __restrict__ out) {
    __shared__ float r[3072];
    int b = blockIdx.x / 3, cc = blockIdx.x % 3;
    int c = cc * 256 + threadIdx.x;
    for (int i = threadIdx.x; i < 3072; i += 256) r[i] = g[b * 3072 + i];
    __syncthreads();
    float acc = b2[c];
    for (int k = 0; k < 3072; ++k) acc += r[k] * W2[k * 768 + c];
    out[((size_t)b * 4096 + 4095) * 768 + c] = last[b * 768 + c] + acc;
}

// ---------------------------------------------------------------------------
extern "C" void kernel_launch(void* const* d_in, const int* in_sizes, int n_in,
                              void* d_out, int out_size, void* d_ws, size_t ws_size,
                              hipStream_t stream) {
    const float* x   = (const float*)d_in[0];
    const float* Wq  = (const float*)d_in[1];
    const float* bq  = (const float*)d_in[2];
    const float* Wkv = (const float*)d_in[3];
    const float* bkv = (const float*)d_in[4];
    const float* Wo  = (const float*)d_in[5];
    const float* bo  = (const float*)d_in[6];
    const float* W1  = (const float*)d_in[7];
    const float* b1  = (const float*)d_in[8];
    const float* W2  = (const float*)d_in[9];
    const float* b2  = (const float*)d_in[10];
    float* out = (float*)d_out;

    char* ws = (char*)d_ws;
    size_t off = 0;
    auto carve = [&](size_t bytes) -> char* {
        char* p = ws + off;
        off = (off + bytes + 255) & ~(size_t)255;
        return p;
    };
    unsigned short* xn   = (unsigned short*)carve(sizeof(unsigned short) * 8ull * 4096 * 768);
    unsigned short* wkvT = (unsigned short*)carve(sizeof(unsigned short) * 1536ull * 768);
    unsigned short* kv   = (unsigned short*)carve(sizeof(unsigned short) * 8ull * 4095 * 1536);
    float* qbuf = (float*)carve(sizeof(float) * 8 * 768);
    float* aout = (float*)carve(sizeof(float) * 8 * 768);
    float* lst  = (float*)carve(sizeof(float) * 8 * 768);
    float* ln2  = (float*)carve(sizeof(float) * 8 * 768);
    float* g    = (float*)carve(sizeof(float) * 8 * 3072);
    (void)ws_size; (void)in_sizes; (void)n_in; (void)out_size;

    const long n_f4 = 8L * 4096 * 768 / 4;             // 6,291,456 float4
    copy_f4_kernel<<<(int)((n_f4 + 255) / 256), 256, 0, stream>>>(
        (const float4*)x, (float4*)out, n_f4);

    ln_rows_kernel<<<4096, 256, 0, stream>>>(x, xn);
    wkvT_bf16_kernel<<<(768 * 1536 + 255) / 256, 256, 0, stream>>>(Wkv, wkvT);
    q_proj_kernel<<<8, 256, 0, stream>>>(xn, Wq, bq, qbuf);

    kv_gemm_kernel<<<dim3(6, 2048), 256, 0, stream>>>(xn, wkvT, bkv, kv);

    attn_kernel<<<128, 256, 0, stream>>>(qbuf, kv, aout);
    proj_ln_kernel<<<8, 256, 0, stream>>>(aout, Wo, bo, x, lst, ln2);
    mlp1_kernel<<<96, 256, 0, stream>>>(ln2, W1, b1, g);
    mlp2_kernel<<<24, 256, 0, stream>>>(g, W2, b2, lst, out);
}
